// PyraformerSSModel_49082886259375
// MI455X (gfx1250) — matrix operand order repaired and checked by reading, not compiled, not run on hardware
//
#include <hip/hip_runtime.h>
#include <math.h>

// ---------------------------------------------------------------------------
// Model constants (Pyraformer forward)
// ---------------------------------------------------------------------------
#define BSZ        4
#define IN_LEN     1024
#define DMODEL     512
#define DBOT       64
#define LTOT       1360          // 1024 + 256 + 64 + 16
#define NHEAD      8
#define DK         64
#define DINNER     2048
#define PYR_LEN    336           // 256 + 64 + 16

typedef __attribute__((ext_vector_type(8)))  float  v8f;
typedef __attribute__((ext_vector_type(16))) __bf16 v16bf;
typedef __attribute__((ext_vector_type(8)))  __bf16 v8bf;

// fp32 -> bf16, round-to-nearest-even
__device__ __forceinline__ __bf16 f2bf(float f) {
    unsigned u = __builtin_bit_cast(unsigned, f);
    unsigned r = u + 0x7FFFu + ((u >> 16) & 1u);
    unsigned short s = (unsigned short)(r >> 16);
    return __builtin_bit_cast(__bf16, s);
}

__device__ __forceinline__ v8bf pack8(float4 a, float4 b) {
    v8bf r;
    r[0] = f2bf(a.x); r[1] = f2bf(a.y); r[2] = f2bf(a.z); r[3] = f2bf(a.w);
    r[4] = f2bf(b.x); r[5] = f2bf(b.y); r[6] = f2bf(b.z); r[7] = f2bf(b.w);
    return r;
}
__device__ __forceinline__ v16bf cat16(v8bf lo, v8bf hi) {
    return __builtin_shufflevector(lo, hi, 0,1,2,3,4,5,6,7,8,9,10,11,12,13,14,15);
}

__device__ __forceinline__ float gelu_tanh(float x) {
    float x3 = x * x * x;
    return 0.5f * x * (1.0f + tanhf(0.7978845608028654f * (x + 0.044715f * x3)));
}

// ---------------------------------------------------------------------------
// Generic WMMA GEMM: out[M,N] = epilogue(A[M,K] @ W[K,N])
// fp32 in / fp32 out, bf16 WMMA core (v_wmma_f32_16x16x32_bf16), fp32 accum.
// Block = 128 threads (4 waves); block computes a 64x64 tile, each wave a
// 16x64 strip. K is consumed in 64-wide stages, double-buffered through
// registers + 2 LDS stages: stage s+1 global loads are issued BEFORE the
// stage-s WMMAs, so VMEM latency overlaps matrix math; one barrier per stage.
// EPI: 0 = +bias, 1 = +bias,GELU(tanh), 2 = +bias,BN,ELU (conv), 3 = none
// Requires: M%64==0, N%64==0, K%64==0 (true for every GEMM in this model).
// ---------------------------------------------------------------------------
struct Stage { float4 a[8]; float4 w[8]; };

__device__ __forceinline__ void issue_loads(const float* __restrict__ A,
                                            const float* __restrict__ W,
                                            int K, int N, int m0, int n0,
                                            int ks0, int t, Stage& st)
{
    int rbase = t >> 3, c4 = t & 7;                 // A: rows rbase+16i, k = c4*4 (+32)
    #pragma unroll
    for (int i = 0; i < 4; ++i) {
        const float* ap = A + (size_t)(m0 + rbase + i * 16) * K + ks0 + c4 * 4;
        st.a[i*2]   = *(const float4*)(ap);
        st.a[i*2+1] = *(const float4*)(ap + 32);
    }
    int n4 = t & 15, kb0 = t >> 4;                  // W: k = kb0+8i (+32), n = n4*4..+3
    #pragma unroll
    for (int i = 0; i < 4; ++i) {
        const float* w0 = W + (size_t)(ks0 + kb0 + i * 8)      * N + n0 + n4 * 4;
        const float* w1 = W + (size_t)(ks0 + kb0 + i * 8 + 32) * N + n0 + n4 * 4;
        st.w[i*2]   = *(const float4*)(w0);
        st.w[i*2+1] = *(const float4*)(w1);
    }
}

__device__ __forceinline__ void store_stage(__bf16 (*sA)[72], __bf16 (*sW)[72],
                                            int t, const Stage& st)
{
    int rbase = t >> 3, c4 = t & 7;
    #pragma unroll
    for (int i = 0; i < 4; ++i) {
        int row = rbase + i * 16;
        #pragma unroll
        for (int half = 0; half < 2; ++half) {
            float4 f = st.a[i*2+half];
            int kk = c4 * 4 + half * 32;
            sA[row][kk+0] = f2bf(f.x); sA[row][kk+1] = f2bf(f.y);
            sA[row][kk+2] = f2bf(f.z); sA[row][kk+3] = f2bf(f.w);
        }
    }
    int n4 = t & 15, kb0 = t >> 4;
    #pragma unroll
    for (int i = 0; i < 4; ++i) {
        #pragma unroll
        for (int half = 0; half < 2; ++half) {
            int kk = kb0 + i * 8 + half * 32;
            float4 f = st.w[i*2+half];
            sW[n4*4+0][kk] = f2bf(f.x); sW[n4*4+1][kk] = f2bf(f.y);
            sW[n4*4+2][kk] = f2bf(f.z); sW[n4*4+3][kk] = f2bf(f.w);
        }
    }
}

template <int EPI>
__global__ void __launch_bounds__(128)
gemm_kernel(const float* __restrict__ A, const float* __restrict__ W,
            const float* __restrict__ bias,
            const float* __restrict__ bng, const float* __restrict__ bnb,
            const float* __restrict__ bnm, const float* __restrict__ bnv,
            float* __restrict__ out, int M, int N, int K)
{
    __shared__ __attribute__((aligned(16))) __bf16 sA[2][64][72]; // [stage][row][k]
    __shared__ __attribute__((aligned(16))) __bf16 sW[2][64][72]; // [stage][n][k]

    const int t    = threadIdx.x;
    const int lane = t & 31;
    const int wave = t >> 5;
    const int hi   = lane >> 4;
    const int ml   = lane & 15;
    const int n0   = blockIdx.x * 64;
    const int m0   = blockIdx.y * 64;
    const int arow = wave * 16 + ml;

    v8f acc[4] = {};

    const int S = K >> 6;                     // 64-wide K stages
    Stage st;
    issue_loads(A, W, K, N, m0, n0, 0, t, st);
    store_stage(sA[0], sW[0], t, st);

    for (int s = 0; s < S; ++s) {
        __syncthreads();                      // stage s resident in LDS
        const int cur  = s & 1;
        const bool more = (s + 1 < S);
        if (more) issue_loads(A, W, K, N, m0, n0, (s + 1) * 64, t, st); // in flight

        #pragma unroll
        for (int ks = 0; ks < 2; ++ks) {      // two 32-wide WMMA K-chunks
            v8bf a0 = *(const v8bf*)&sA[cur][arow][ks * 32 + hi * 8];
            v8bf a1 = *(const v8bf*)&sA[cur][arow][ks * 32 + 16 + hi * 8];
            v16bf a = cat16(a0, a1);
            #pragma unroll
            for (int nt = 0; nt < 4; ++nt) {
                v8bf b0 = *(const v8bf*)&sW[cur][nt * 16 + ml][ks * 32 + hi * 8];
                v8bf b1 = *(const v8bf*)&sW[cur][nt * 16 + ml][ks * 32 + 16 + hi * 8];
                acc[nt] = __builtin_amdgcn_wmma_f32_16x16x32_bf16(
                    false, a, false, cat16(b0, b1), (short)0, acc[nt], false, false);
            }
        }
        if (more) store_stage(sA[cur ^ 1], sW[cur ^ 1], t, st);
    }

    // --- epilogue + store: lane ml = column, VGPR r -> row r + 8*hi ---
    #pragma unroll
    for (int nt = 0; nt < 4; ++nt) {
        #pragma unroll
        for (int r = 0; r < 8; ++r) {
            int m = m0 + wave * 16 + r + 8 * hi;
            int n = n0 + nt * 16 + ml;
            float v = acc[nt][r];
            if (EPI == 0) {
                v += bias[n];
            } else if (EPI == 1) {
                v = gelu_tanh(v + bias[n]);
            } else if (EPI == 2) {
                v += bias[n];
                v = (v - bnm[n]) * rsqrtf(bnv[n] + 1e-5f) * bng[n] + bnb[n];
                v = (v > 0.0f) ? v : expm1f(v);              // ELU
            }
            out[(size_t)m * N + n] = v;
        }
    }
}

// ---------------------------------------------------------------------------
// Flash attention: one wave per (b, h, 16-row query block).
// S = (Q K^T) * 0.125 with mask -> online softmax -> O += P V, all via WMMA.
// Q/K/V are (B*L, 512) fp32, head h occupies columns [h*64, h*64+64).
// Key loop step = 32 (two 16-key n-tiles); L = 1360 = 42*32 + 16 tail.
// ---------------------------------------------------------------------------
__global__ void __launch_bounds__(32)
attn_kernel(const float* __restrict__ Q, const float* __restrict__ Km,
            const float* __restrict__ Vm, const unsigned char* __restrict__ mask,
            float* __restrict__ O)
{
    __shared__ __attribute__((aligned(16))) __bf16 pscr[16][40]; // P tile 16x32 bf16
    __shared__ __attribute__((aligned(16))) __bf16 vscr[64][40]; // V tile transposed [dim][key]

    const int lane  = threadIdx.x & 31;
    const int hi    = lane >> 4;
    const int ml    = lane & 15;
    const int qblk  = blockIdx.x;          // 0..84
    const int bh    = blockIdx.y;          // 0..B*H-1
    const int b     = bh >> 3;
    const int h     = bh & 7;
    const int qbase = qblk * 16;
    const size_t bL = (size_t)b * LTOT;

    // Q A-fragments for K-dim chunks d in [0,32) and [32,64); constant over keys
    v16bf qa[2];
    {
        const float* qrow = Q + (bL + qbase + ml) * DMODEL + h * DK;
        #pragma unroll
        for (int ks = 0; ks < 2; ++ks) {
            float4 f0a = *(const float4*)(qrow + ks*32 + hi*8);
            float4 f0b = *(const float4*)(qrow + ks*32 + hi*8 + 4);
            float4 f1a = *(const float4*)(qrow + ks*32 + 16 + hi*8);
            float4 f1b = *(const float4*)(qrow + ks*32 + 16 + hi*8 + 4);
            qa[ks] = cat16(pack8(f0a, f0b), pack8(f1a, f1b));
        }
    }

    float mrun[8], srun[8];
    v8f o[4] = {};
    #pragma unroll
    for (int r = 0; r < 8; ++r) { mrun[r] = -3.0e38f; srun[r] = 0.0f; }

    for (int kb = 0; kb < LTOT; kb += 32) {
        // prefetch next key block's K and V rows (global_prefetch_b8 path)
        if (kb + 32 + lane < LTOT) {
            __builtin_prefetch(Km + (bL + kb + 32 + lane) * DMODEL + h * DK, 0, 0);
            __builtin_prefetch(Vm + (bL + kb + 32 + lane) * DMODEL + h * DK, 0, 0);
        }

        // ---- scores S (16x32) = Q K^T : two n-tiles, two K chunks each ----
        float sm[2][8];
        #pragma unroll
        for (int nt = 0; nt < 2; ++nt) {
            if (kb + nt * 16 < LTOT) {                       // 16-key block validity
                int key = kb + nt * 16 + ml;
                const float* krow = Km + (bL + key) * DMODEL + h * DK;
                v8f z = {};
                #pragma unroll
                for (int ks = 0; ks < 2; ++ks) {
                    const float* kp = krow + ks * 32 + hi * 8; // d = hi*8..+7 and 16+hi*8..+7
                    float4 c0 = *(const float4*)(kp);
                    float4 c1 = *(const float4*)(kp + 4);
                    float4 c2 = *(const float4*)(kp + 16);
                    float4 c3 = *(const float4*)(kp + 20);
                    z = __builtin_amdgcn_wmma_f32_16x16x32_bf16(
                        false, qa[ks], false, cat16(pack8(c0, c1), pack8(c2, c3)),
                        (short)0, z, false, false);
                }
                #pragma unroll
                for (int r = 0; r < 8; ++r) {
                    int qg = qbase + r + 8 * hi;
                    float v = z[r] * 0.125f;                  // 1/sqrt(64)
                    if (mask[(size_t)qg * LTOT + key]) v = -1.0e9f;
                    sm[nt][r] = v;
                }
            } else {
                #pragma unroll
                for (int r = 0; r < 8; ++r) sm[nt][r] = -3.0e38f; // OOB -> exp()==0
            }
        }

        // ---- online softmax update (row stats live in 16-lane halves) ----
        float pp[2][8];
        #pragma unroll
        for (int r = 0; r < 8; ++r) {
            float rm = fmaxf(sm[0][r], sm[1][r]);
            for (int off = 1; off < 16; off <<= 1) rm = fmaxf(rm, __shfl_xor(rm, off));
            float nm    = fmaxf(mrun[r], rm);
            float alpha = __expf(mrun[r] - nm);
            float p0 = __expf(sm[0][r] - nm);
            float p1 = __expf(sm[1][r] - nm);
            float rs = p0 + p1;
            for (int off = 1; off < 16; off <<= 1) rs += __shfl_xor(rs, off);
            srun[r] = srun[r] * alpha + rs;
            mrun[r] = nm;
            pp[0][r] = p0; pp[1][r] = p1;
            #pragma unroll
            for (int nt = 0; nt < 4; ++nt) o[nt][r] *= alpha;
        }

        __syncthreads();
        // P (C-frag layout) -> LDS so it can be re-read in A-frag layout
        #pragma unroll
        for (int r = 0; r < 8; ++r) {
            pscr[r + 8 * hi][ml]      = f2bf(pp[0][r]);
            pscr[r + 8 * hi][16 + ml] = f2bf(pp[1][r]);
        }
        // V block (32 keys x 64 dims) -> LDS transposed [dim][key], bf16
        {
            int key = kb + lane;
            if (key < LTOT) {
                const float* vrow = Vm + (bL + key) * DMODEL + h * DK;
                #pragma unroll
                for (int c4 = 0; c4 < 16; ++c4) {
                    float4 f = *(const float4*)(vrow + c4 * 4);
                    vscr[c4*4+0][lane] = f2bf(f.x);
                    vscr[c4*4+1][lane] = f2bf(f.y);
                    vscr[c4*4+2][lane] = f2bf(f.z);
                    vscr[c4*4+3][lane] = f2bf(f.w);
                }
            } else {
                #pragma unroll
                for (int d = 0; d < 64; ++d) vscr[d][lane] = f2bf(0.0f);
            }
        }
        __syncthreads();

        // ---- O += P V : one A-frag (P 16x32), four B-frags (V 32x16) ----
        v8bf plo = *(const v8bf*)&pscr[ml][hi * 8];
        v8bf phi = *(const v8bf*)&pscr[ml][16 + hi * 8];
        v16bf pa = cat16(plo, phi);
        #pragma unroll
        for (int nt = 0; nt < 4; ++nt) {
            v8bf b0 = *(const v8bf*)&vscr[nt * 16 + ml][hi * 8];
            v8bf b1 = *(const v8bf*)&vscr[nt * 16 + ml][16 + hi * 8];
            o[nt] = __builtin_amdgcn_wmma_f32_16x16x32_bf16(
                false, pa, false, cat16(b0, b1), (short)0, o[nt], false, false);
        }
    }

    // ---- normalize and store (column h*64 + d of the (B*L,512) output) ----
    #pragma unroll
    for (int nt = 0; nt < 4; ++nt)
        #pragma unroll
        for (int r = 0; r < 8; ++r) {
            int m = qbase + r + 8 * hi;
            O[(bL + m) * DMODEL + h * DK + nt * 16 + ml] = o[nt][r] / srun[r];
        }
}

// ---------------------------------------------------------------------------
// LayerNorm over concat([x, pyr], seq axis): one wave per row of 512.
// ---------------------------------------------------------------------------
__global__ void __launch_bounds__(256)
ln_concat_kernel(const float* __restrict__ x, const float* __restrict__ pyr,
                 const float* __restrict__ g, const float* __restrict__ bt,
                 float* __restrict__ seq)
{
    int wave = threadIdx.x >> 5, lane = threadIdx.x & 31;
    int row = blockIdx.x * 8 + wave;                   // < B*LTOT
    int bi = row / LTOT, r = row % LTOT;
    const float* src = (r < IN_LEN)
        ? x   + ((size_t)bi * IN_LEN  + r)            * DMODEL
        : pyr + ((size_t)bi * PYR_LEN + (r - IN_LEN)) * DMODEL;

    float vb[16]; float s = 0.0f;
    #pragma unroll
    for (int i = 0; i < 16; ++i) { float v = src[lane + i * 32]; vb[i] = v; s += v; }
    for (int off = 16; off > 0; off >>= 1) s += __shfl_xor(s, off);
    float mu = s * (1.0f / DMODEL);
    float q = 0.0f;
    #pragma unroll
    for (int i = 0; i < 16; ++i) { float d = vb[i] - mu; q += d * d; }
    for (int off = 16; off > 0; off >>= 1) q += __shfl_xor(q, off);
    float inv = rsqrtf(q * (1.0f / DMODEL) + 1e-5f);
    #pragma unroll
    for (int i = 0; i < 16; ++i) {
        int c = lane + i * 32;
        seq[(size_t)row * DMODEL + c] = (vb[i] - mu) * inv * g[c] + bt[c];
    }
}

// seq = LN(a + seq_in); safe in-place on seq.
__global__ void __launch_bounds__(256)
ln_residual_kernel(const float* __restrict__ a, const float* __restrict__ seq_in,
                   const float* __restrict__ g, const float* __restrict__ bt,
                   float* __restrict__ seq_out)
{
    int wave = threadIdx.x >> 5, lane = threadIdx.x & 31;
    size_t row = (size_t)blockIdx.x * 8 + wave;
    const float* pa = a      + row * DMODEL;
    const float* ps = seq_in + row * DMODEL;
    float vb[16]; float s = 0.0f;
    #pragma unroll
    for (int i = 0; i < 16; ++i) {
        int c = lane + i * 32;
        float v = pa[c] + ps[c]; vb[i] = v; s += v;
    }
    for (int off = 16; off > 0; off >>= 1) s += __shfl_xor(s, off);
    float mu = s * (1.0f / DMODEL);
    float q = 0.0f;
    #pragma unroll
    for (int i = 0; i < 16; ++i) { float d = vb[i] - mu; q += d * d; }
    for (int off = 16; off > 0; off >>= 1) q += __shfl_xor(q, off);
    float inv = rsqrtf(q * (1.0f / DMODEL) + 1e-5f);
    #pragma unroll
    for (int i = 0; i < 16; ++i) {
        int c = lane + i * 32;
        seq_out[row * DMODEL + c] = (vb[i] - mu) * inv * g[c] + bt[c];
    }
}

// conv weight (O=64, I=64, T=4) -> row-major (K=256, N=64) with kk = t*64 + i
__global__ void __launch_bounds__(256)
convw_reorder_kernel(const float* __restrict__ w, float* __restrict__ wf)
{
    int tid = blockIdx.x * 256 + threadIdx.x;   // 16384
    int o = tid & 63, kk = tid >> 6;
    int tt = kk >> 6, ii = kk & 63;
    wf[tid] = w[o * 256 + ii * 4 + tt];
}

// concat conv outputs along seq axis -> (B*336, 64)
__global__ void __launch_bounds__(256)
pyrcat_kernel(const float* __restrict__ c1, const float* __restrict__ c2,
              const float* __restrict__ c3, float* __restrict__ pin)
{
    int tid = blockIdx.x * 256 + threadIdx.x;   // B*336*64
    int c = tid & 63, rr = tid >> 6;
    int bi = rr / PYR_LEN, j = rr % PYR_LEN;
    float v;
    if (j < 256)      v = c1[((size_t)bi * 256 + j)         * 64 + c];
    else if (j < 320) v = c2[((size_t)bi * 64  + (j - 256)) * 64 + c];
    else              v = c3[((size_t)bi * 16  + (j - 320)) * 64 + c];
    pin[tid] = v;
}

// out[b][i][j*512 + c] = seq[b][idx[i][j]][c], 128-bit copies
__global__ void __launch_bounds__(256)
gather_kernel(const float* __restrict__ seq, const int* __restrict__ idx,
              float* __restrict__ out)
{
    size_t tid = (size_t)blockIdx.x * 256 + threadIdx.x;   // 2,097,152 float4s
    int c4 = tid & 127;
    int j  = (tid >> 7) & 3;
    int i  = (tid >> 9) & 1023;
    int b  = (int)(tid >> 19);
    int row = idx[i * 4 + j];
    float4 v = *(const float4*)(seq + ((size_t)b * LTOT + row) * DMODEL + c4 * 4);
    *(float4*)(out + (tid << 2)) = v;
}

// ---------------------------------------------------------------------------
// Host side
// ---------------------------------------------------------------------------
static void launch_gemm(int epi, const float* A, const float* W, const float* bias,
                        const float* bng, const float* bnb, const float* bnm,
                        const float* bnv, float* out, int M, int N, int K,
                        hipStream_t s)
{
    dim3 grid(N / 64, M / 64), blk(128);
    switch (epi) {
    case 0: gemm_kernel<0><<<grid, blk, 0, s>>>(A, W, bias, bng, bnb, bnm, bnv, out, M, N, K); break;
    case 1: gemm_kernel<1><<<grid, blk, 0, s>>>(A, W, bias, bng, bnb, bnm, bnv, out, M, N, K); break;
    case 2: gemm_kernel<2><<<grid, blk, 0, s>>>(A, W, bias, bng, bnb, bnm, bnv, out, M, N, K); break;
    default: gemm_kernel<3><<<grid, blk, 0, s>>>(A, W, bias, bng, bnb, bnm, bnv, out, M, N, K); break;
    }
}

struct Params {
    const float *down_w, *down_b, *up_w, *up_b, *norm_g, *norm_b;
    const float *cw[3], *cb[3], *bng[3], *bnb[3], *bnm[3], *bnv[3];
    const float *wq[2], *wk[2], *wv[2], *fcw[2], *fcb[2];
    const float *l1g[2], *l1b[2], *w1[2], *b1[2], *w2[2], *b2[2], *l2g[2], *l2b[2];
};

extern "C" void kernel_launch(void* const* d_in, const int* in_sizes, int n_in,
                              void* d_out, int out_size, void* d_ws, size_t ws_size,
                              hipStream_t stream)
{
    const float* x = (const float*)d_in[0];
    const unsigned char* mask = (const unsigned char*)d_in[n_in - 2]; // (1360,1360) bool
    const int* indexes = (const int*)d_in[n_in - 1];                  // (1024,4) int32

    // Map param leaves. Two candidate flattening orders:
    //  (a) JAX tree_flatten (dict keys sorted alphabetically)  -> in_sizes[1]==64 (convs[0].b)
    //  (b) insertion order                                     -> in_sizes[1]==32768 (down_w)
    Params P;
    int p = 1;
    auto F = [&](void) -> const float* { return (const float*)d_in[p++]; };
    if (in_sizes[1] == 64) { // alphabetical
        for (int c = 0; c < 3; ++c) {
            P.cb[c] = F(); P.bnb[c] = F(); P.bng[c] = F(); P.bnm[c] = F(); P.bnv[c] = F(); P.cw[c] = F();
        }
        P.down_b = F(); P.down_w = F();
        for (int l = 0; l < 2; ++l) {
            P.b1[l] = F(); P.b2[l] = F(); P.fcb[l] = F(); P.fcw[l] = F();
            P.l1b[l] = F(); P.l1g[l] = F(); P.l2b[l] = F(); P.l2g[l] = F();
            P.w1[l] = F(); P.w2[l] = F(); P.wk[l] = F(); P.wq[l] = F(); P.wv[l] = F();
        }
        P.norm_b = F(); P.norm_g = F(); P.up_b = F(); P.up_w = F();
    } else {                 // insertion order
        P.down_w = F(); P.down_b = F();
        for (int c = 0; c < 3; ++c) {
            P.cw[c] = F(); P.cb[c] = F(); P.bng[c] = F(); P.bnb[c] = F(); P.bnm[c] = F(); P.bnv[c] = F();
        }
        P.up_w = F(); P.up_b = F(); P.norm_g = F(); P.norm_b = F();
        for (int l = 0; l < 2; ++l) {
            P.wq[l] = F(); P.wk[l] = F(); P.wv[l] = F(); P.fcw[l] = F(); P.fcb[l] = F();
            P.l1g[l] = F(); P.l1b[l] = F(); P.w1[l] = F(); P.b1[l] = F();
            P.w2[l] = F(); P.b2[l] = F(); P.l2g[l] = F(); P.l2b[l] = F();
        }
    }

    // ---- workspace layout (floats), ~72 MB total ----
    float* ws = (float*)d_ws;
    size_t off = 0;
    auto alloc = [&](size_t n) { float* q = ws + off; off += n; return q; };
    float* t    = alloc((size_t)BSZ * IN_LEN * DBOT);        // 262144
    float* c1   = alloc((size_t)BSZ * 256 * DBOT);           // 65536
    float* c2   = alloc((size_t)BSZ * 64  * DBOT);           // 16384
    float* c3   = alloc((size_t)BSZ * 16  * DBOT);           // 4096
    float* wf0  = alloc(256 * 64);
    float* wf1  = alloc(256 * 64);
    float* wf2  = alloc(256 * 64);
    float* pin  = alloc((size_t)BSZ * PYR_LEN * DBOT);       // 86016
    float* pyr  = alloc((size_t)BSZ * PYR_LEN * DMODEL);     // 688128
    const size_t SEQN = (size_t)BSZ * LTOT * DMODEL;         // 2785280
    float* seq  = alloc(SEQN);
    float* qb   = alloc(SEQN);
    float* kb   = alloc(SEQN);
    float* vb   = alloc(SEQN);
    float* ao   = alloc(SEQN);
    float* fco  = alloc(SEQN);
    float* ffi  = qb;   // FFN intermediate (B*L, 2048) overlays q/k/v/ao exactly

    const int ML = BSZ * LTOT;   // 5440

    // 1) down-projection: (4096,512) @ (512,64) + b
    launch_gemm(0, x, P.down_w, P.down_b, 0, 0, 0, 0, t, BSZ * IN_LEN, DBOT, DMODEL, stream);

    // 2) conv weights -> (K=256, N=64) row-major
    convw_reorder_kernel<<<64, 256, 0, stream>>>(P.cw[0], wf0);
    convw_reorder_kernel<<<64, 256, 0, stream>>>(P.cw[1], wf1);
    convw_reorder_kernel<<<64, 256, 0, stream>>>(P.cw[2], wf2);

    // 3) stride-4 convs == reshaped GEMMs (+bias+BN+ELU epilogue)
    launch_gemm(2, t,  wf0, P.cb[0], P.bng[0], P.bnb[0], P.bnm[0], P.bnv[0], c1, BSZ * 256, 64, 256, stream);
    launch_gemm(2, c1, wf1, P.cb[1], P.bng[1], P.bnb[1], P.bnm[1], P.bnv[1], c2, BSZ * 64,  64, 256, stream);
    launch_gemm(2, c2, wf2, P.cb[2], P.bng[2], P.bnb[2], P.bnm[2], P.bnv[2], c3, BSZ * 16,  64, 256, stream);

    // 4) pyramid concat + up-projection
    pyrcat_kernel<<<(BSZ * PYR_LEN * 64) / 256, 256, 0, stream>>>(c1, c2, c3, pin);
    launch_gemm(0, pin, P.up_w, P.up_b, 0, 0, 0, 0, pyr, BSZ * PYR_LEN, DMODEL, DBOT, stream);

    // 5) seq = LN(concat([x, pyr]))
    ln_concat_kernel<<<ML / 8, 256, 0, stream>>>(x, pyr, P.norm_g, P.norm_b, seq);

    // 6) transformer layers
    for (int l = 0; l < 2; ++l) {
        launch_gemm(3, seq, P.wq[l], 0, 0, 0, 0, 0, qb, ML, DMODEL, DMODEL, stream);
        launch_gemm(3, seq, P.wk[l], 0, 0, 0, 0, 0, kb, ML, DMODEL, DMODEL, stream);
        launch_gemm(3, seq, P.wv[l], 0, 0, 0, 0, 0, vb, ML, DMODEL, DMODEL, stream);

        attn_kernel<<<dim3(LTOT / 16, BSZ * NHEAD), 32, 0, stream>>>(qb, kb, vb, mask, ao);

        launch_gemm(0, ao, P.fcw[l], P.fcb[l], 0, 0, 0, 0, fco, ML, DMODEL, DMODEL, stream);
        ln_residual_kernel<<<ML / 8, 256, 0, stream>>>(fco, seq, P.l1g[l], P.l1b[l], seq);

        launch_gemm(1, seq, P.w1[l], P.b1[l], 0, 0, 0, 0, ffi, ML, DINNER, DMODEL, stream); // GELU
        launch_gemm(0, ffi, P.w2[l], P.b2[l], 0, 0, 0, 0, fco, ML, DMODEL, DINNER, stream);
        ln_residual_kernel<<<ML / 8, 256, 0, stream>>>(fco, seq, P.l2g[l], P.l2b[l], seq);
    }

    // 7) final gather -> (B, 1024, 4*512)
    gather_kernel<<<8192, 256, 0, stream>>>(seq, indexes, (float*)d_out);
    (void)in_sizes; (void)out_size; (void)ws_size;
}
